// HierarchicalQueryMatcher_83958020702582
// MI455X (gfx1250) — matrix operand order repaired and checked by reading, not compiled, not run on hardware
//
#include <hip/hip_runtime.h>
#include <hip/hip_bf16.h>

typedef __attribute__((ext_vector_type(16))) _Float16 v16h;
typedef __attribute__((ext_vector_type(8)))  float    v8f;

#define NQ 2048
#define NW 64
#define DD 512
#define GSTRIDE 520   // halves per row in LDS G tile (4-bank skew per row)
#define HSTRIDE 516   // floats per row in LDS H staging tile

// dynamic LDS layout (all section sizes 16B-aligned)
#define GBYTES   (64 * GSTRIDE * 2)                 // 66560
#define HBYTES   (16 * HSTRIDE * 4)                 // 33024
#define SBYTES   (2 * 64 * 4)                       // 512 (mu,istd)
#define EBYTES   (3 * DD * 4)                       // 6144 (ext, lw, lb)
#define SMEM_BYTES (GBYTES + HBYTES + SBYTES + EBYTES)

__device__ __forceinline__ float wred32(float v) {
#pragma unroll
  for (int off = 16; off >= 1; off >>= 1) v += __shfl_xor(v, off, 32);
  return v;
}

// ---- per-row sum / sumsq of q ----
__global__ void prep_q_kernel(const float* __restrict__ q,
                              float* __restrict__ sq, float* __restrict__ sqq) {
  __shared__ float r1[256], r2[256];
  const int n = blockIdx.x, t = threadIdx.x;
  float a = q[n * DD + t], b = q[n * DD + t + 256];
  r1[t] = a + b;
  r2[t] = a * a + b * b;
  __syncthreads();
  for (int s = 128; s > 0; s >>= 1) {
    if (t < s) { r1[t] += r1[t + s]; r2[t] += r2[t + s]; }
    __syncthreads();
  }
  if (t == 0) { sq[n] = r1[0]; sqq[n] = r2[0]; }
}

// ---- per-row sum / sumsq of c, plus unit-normalized c ----
__global__ void prep_c_kernel(const float* __restrict__ c,
                              float* __restrict__ sc, float* __restrict__ scc,
                              float* __restrict__ cu) {
  __shared__ float r1[256], r2[256];
  __shared__ float inv;
  const int w = blockIdx.x, t = threadIdx.x;
  float a = c[w * DD + t], b = c[w * DD + t + 256];
  r1[t] = a + b;
  r2[t] = a * a + b * b;
  __syncthreads();
  for (int s = 128; s > 0; s >>= 1) {
    if (t < s) { r1[t] += r1[t + s]; r2[t] += r2[t + s]; }
    __syncthreads();
  }
  if (t == 0) {
    sc[w] = r1[0];
    scc[w] = r2[0];
    inv = 1.0f / fmaxf(sqrtf(r2[0]), 1e-12f);
  }
  __syncthreads();
  cu[w * DD + t] = a * inv;
  cu[w * DD + t + 256] = b * inv;
}

// ---- column reductions of w1 ----
__global__ void prep_cols_kernel(const float* __restrict__ w1,
                                 const float* __restrict__ ln1w,
                                 const float* __restrict__ ln1b,
                                 const float* __restrict__ b1,
                                 float* __restrict__ colsum, float* __restrict__ cb) {
  const int k = blockIdx.x * 256 + threadIdx.x;
  float a = 0.f, b = 0.f;
  for (int j = 0; j < 2 * DD; ++j) {
    float wv = w1[j * DD + k];
    a += ln1w[j] * wv;
    b += ln1b[j] * wv;
  }
  colsum[k] = a;
  cb[k] = b + b1[k];
}

// ---- w2 -> f16, transposed to [j][k] ----
__global__ void conv_w2_kernel(const float* __restrict__ w2, _Float16* __restrict__ w2t) {
  const int idx = blockIdx.x * 256 + threadIdx.x; // k*512 + j
  const int k = idx >> 9, j = idx & (DD - 1);
  w2t[j * DD + k] = (_Float16)w2[idx];
}

// ---- fp32 tiled GEMM: Out[m,k] = sum_j X[m,j] * ln1w[woff+j] * w1[(woff+j),k] ----
__global__ void gemm_ln_f32(const float* __restrict__ X, const float* __restrict__ w1,
                            const float* __restrict__ ln1w, int woff,
                            float* __restrict__ Out) {
  __shared__ float as[16][16];
  __shared__ float bs[16][17];
  const int tx = threadIdx.x & 15, ty = threadIdx.x >> 4;
  const int m = blockIdx.x * 16 + ty;
  const int nc = blockIdx.y * 16 + tx;
  float acc = 0.f;
  for (int k0 = 0; k0 < DD; k0 += 16) {
    as[ty][tx] = X[m * DD + k0 + tx];
    const int j = woff + k0 + ty;
    bs[ty][tx] = ln1w[j] * w1[j * DD + nc];
    __syncthreads();
#pragma unroll
    for (int p = 0; p < 16; ++p) acc += as[ty][p] * bs[p][tx];
    __syncthreads();
  }
  Out[m * DD + nc] = acc;
}

// ---- fused main kernel: one block = one n x all 64 w ----
__global__ void __launch_bounds__(256) matcher_main(
    const float* __restrict__ A, const float* __restrict__ Bc,
    const float* __restrict__ colsum, const float* __restrict__ cb,
    const float* __restrict__ sq, const float* __restrict__ sqq,
    const float* __restrict__ sc, const float* __restrict__ scc,
    const _Float16* __restrict__ w2t, const float* __restrict__ b2,
    const float* __restrict__ q, const float* __restrict__ ln2w,
    const float* __restrict__ ln2b, const float* __restrict__ cu,
    float* __restrict__ out) {
  extern __shared__ char smem[];
  _Float16* gG     = (_Float16*)smem;                          // 64 x GSTRIDE halves
  float*    H16    = (float*)(smem + GBYTES);                  // 16 x HSTRIDE floats
  float*    mu_s   = (float*)(smem + GBYTES + HBYTES);         // 64
  float*    istd_s = mu_s + 64;                                // 64
  float*    ext    = (float*)(smem + GBYTES + HBYTES + SBYTES);// b2[j]+q[n,j]
  float*    lw     = ext + DD;                                 // ln2_w
  float*    lb     = lw + DD;                                  // ln2_b

  const int tid = threadIdx.x;
  const int n = blockIdx.x;

  if (tid < 64) {
    float mu = (sq[n] + sc[tid]) * (1.0f / (2.0f * DD));
    float var = (sqq[n] + scc[tid]) * (1.0f / (2.0f * DD)) - mu * mu;
    mu_s[tid] = mu;
    istd_s[tid] = rsqrtf(var + 1e-5f);
  }
  // stage epilogue vectors once per block
  for (int j = tid; j < DD; j += 256) {
    ext[j] = b2[j] + q[n * DD + j];
    lw[j] = ln2w[j];
    lb[j] = ln2b[j];
  }
  __syncthreads();

  // Phase 1: gelu'd activation tile for all 64 rows (w)
  for (int idx = tid; idx < 64 * DD; idx += 256) {
    const int r = idx >> 9;
    const int k = idx & (DD - 1);
    float pre = istd_s[r] * (A[n * DD + k] + Bc[r * DD + k] - mu_s[r] * colsum[k]) + cb[k];
    float g = 0.5f * pre * (1.0f + erff(pre * 0.70710678118654752f));
    gG[r * GSTRIDE + k] = (_Float16)g;
  }
  __syncthreads();

  // Phase 2: WMMA GEMM (64x512 @ 512x512)
  const int lane = tid & 31;
  const int wv = tid >> 5;
  const int j0 = wv << 6;
  const int row = lane & 15;
  const int hi = lane >> 4;

  v8f acc[16];                   // acc[jt*4 + rt]
#pragma unroll
  for (int i = 0; i < 16; ++i) acc[i] = (v8f){};

  const unsigned* gGu = (const unsigned*)gG;
  const unsigned* w2u = (const unsigned*)w2t;

  for (int kt = 0; kt < DD / 32; ++kt) {
    const int kbase = kt * 32;
    union Frag { v16h h; unsigned u[8]; uint4 q[2]; };
    Frag af[4];

    // A fragments for the 4 row-tiles: 16B-aligned -> ds_load_b128
#pragma unroll
    for (int rt = 0; rt < 4; ++rt) {
      const int abase = ((rt * 16 + row) * GSTRIDE + kbase + hi * 8) >> 1;
      const uint4* pa = (const uint4*)(gGu + abase);
      af[rt].q[0] = pa[0];
      af[rt].q[1] = pa[2];   // +8 dwords (K = kb+16..23)
    }

    const int bbase = ((j0 + row) * DD + kbase + hi * 8) >> 1;
    __builtin_prefetch(w2u + bbase + 16, 0, 0);   // next kt slab -> global_prefetch

#pragma unroll
    for (int jt = 0; jt < 4; ++jt) {
      Frag bf;
      const uint4* pb = (const uint4*)(w2u + bbase + jt * (16 * DD >> 1));
      bf.q[0] = pb[0];
      bf.q[1] = pb[2];
#pragma unroll
      for (int rt = 0; rt < 4; ++rt) {
        acc[jt * 4 + rt] = __builtin_amdgcn_wmma_f32_16x16x32_f16(
            false, af[rt].h, false, bf.h, (short)0, acc[jt * 4 + rt], false, false);
      }
    }
  }

  // Phases 3/4: four 16-row rounds through the fp32 staging tile
  for (int rt = 0; rt < 4; ++rt) {
#pragma unroll
    for (int jt = 0; jt < 4; ++jt) {
#pragma unroll
      for (int i = 0; i < 8; ++i) {
        const int r = i + hi * 8;
        H16[r * HSTRIDE + j0 + jt * 16 + row] = acc[jt * 4 + rt][i];
      }
    }
    __syncthreads();

    for (int rr = 0; rr < 2; ++rr) {
      const int r = wv + rr * 8;
      const int w = rt * 16 + r;
      float4 y4[4];
      float s1 = 0.f, s2 = 0.f;
#pragma unroll
      for (int c = 0; c < 4; ++c) {
        const int j = 4 * lane + 128 * c;
        float4 h = *(const float4*)&H16[r * HSTRIDE + j];
        float4 e = *(const float4*)&ext[j];
        float4 y;
        y.x = h.x + e.x; y.y = h.y + e.y; y.z = h.z + e.z; y.w = h.w + e.w;
        y4[c] = y;
        s1 += y.x + y.y + y.z + y.w;
        s2 += y.x * y.x + y.y * y.y + y.z * y.z + y.w * y.w;
      }
      s1 = wred32(s1);
      s2 = wred32(s2);
      const float mu = s1 * (1.0f / DD);
      const float istd = rsqrtf(s2 * (1.0f / DD) - mu * mu + 1e-5f);
      float zz = 0.f, zc = 0.f;
#pragma unroll
      for (int c = 0; c < 4; ++c) {
        const int j = 4 * lane + 128 * c;
        float4 w4 = *(const float4*)&lw[j];
        float4 b4 = *(const float4*)&lb[j];
        float4 c4 = *(const float4*)&cu[w * DD + j];
        float4 y = y4[c];
        float zx = (y.x - mu) * istd * w4.x + b4.x;
        float zy = (y.y - mu) * istd * w4.y + b4.y;
        float zzv = (y.z - mu) * istd * w4.z + b4.z;
        float zw = (y.w - mu) * istd * w4.w + b4.w;
        zz += zx * zx + zy * zy + zzv * zzv + zw * zw;
        zc += zx * c4.x + zy * c4.y + zzv * c4.z + zw * c4.w;
      }
      zz = wred32(zz);
      zc = wred32(zc);
      if (lane == 0) out[n * NW + w] = 16.0f * zc / fmaxf(sqrtf(zz), 1e-12f);
    }
    __syncthreads();
  }
}

extern "C" void kernel_launch(void* const* d_in, const int* in_sizes, int n_in,
                              void* d_out, int out_size, void* d_ws, size_t ws_size,
                              hipStream_t stream) {
  (void)in_sizes; (void)n_in; (void)out_size; (void)ws_size;
  const float* q    = (const float*)d_in[0];
  const float* cm   = (const float*)d_in[1];
  const float* ln1w = (const float*)d_in[2];
  const float* ln1b = (const float*)d_in[3];
  const float* w1   = (const float*)d_in[4];
  const float* b1   = (const float*)d_in[5];
  const float* w2   = (const float*)d_in[6];
  const float* b2   = (const float*)d_in[7];
  const float* ln2w = (const float*)d_in[8];
  const float* ln2b = (const float*)d_in[9];
  float* out = (float*)d_out;

  char* ws = (char*)d_ws;
  auto alloc = [&](size_t bytes) -> char* {
    char* p = ws;
    ws += (bytes + 255) & ~(size_t)255;
    return p;
  };
  float*    A      = (float*)alloc((size_t)NQ * DD * 4);
  float*    Bc     = (float*)alloc((size_t)NW * DD * 4);
  float*    cu     = (float*)alloc((size_t)NW * DD * 4);
  float*    colsum = (float*)alloc(DD * 4);
  float*    cbv    = (float*)alloc(DD * 4);
  float*    sq     = (float*)alloc(NQ * 4);
  float*    sqq    = (float*)alloc(NQ * 4);
  float*    sc     = (float*)alloc(NW * 4);
  float*    scc    = (float*)alloc(NW * 4);
  _Float16* w2t    = (_Float16*)alloc((size_t)DD * DD * 2);

  prep_q_kernel<<<NQ, 256, 0, stream>>>(q, sq, sqq);
  prep_c_kernel<<<NW, 256, 0, stream>>>(cm, sc, scc, cu);
  prep_cols_kernel<<<2, 256, 0, stream>>>(w1, ln1w, ln1b, b1, colsum, cbv);
  conv_w2_kernel<<<(DD * DD) / 256, 256, 0, stream>>>(w2, w2t);
  gemm_ln_f32<<<dim3(NQ / 16, DD / 16), 256, 0, stream>>>(q, w1, ln1w, 0, A);
  gemm_ln_f32<<<dim3(NW / 16, DD / 16), 256, 0, stream>>>(cm, w1, ln1w, DD, Bc);
  matcher_main<<<NQ, 256, SMEM_BYTES, stream>>>(
      A, Bc, colsum, cbv, sq, sqq, sc, scc, w2t, b2, q, ln2w, ln2b, cu, out);
}
